// GRAN_52467320487971
// MI455X (gfx1250) — matrix operand order repaired and checked by reading, not compiled, not run on hardware
//
#include <hip/hip_runtime.h>
#include <math.h>

// ---------------- problem constants (match reference) ----------------
#define NN      16000
#define DD      128
#define HH      2
#define CC      128
#define EE      256000
#define II      512
#define LL      2
#define BB      64
#define NTYPES  10
#define HC      (HH*CC)        // 256
#define CAT     (HC*3 + CC)    // 896: [q|k|v|skip]

// ---------------- helpers ----------------
typedef __attribute__((ext_vector_type(16))) __bf16 v16bf;
typedef __attribute__((ext_vector_type(8)))  float  v8f;

__device__ __forceinline__ unsigned short f2bf(float f) {
    unsigned int u = __float_as_uint(f);
    u += 0x7FFFu + ((u >> 16) & 1u);   // round-to-nearest-even
    return (unsigned short)(u >> 16);
}

__device__ __forceinline__ void atomicMaxFloat(float* addr, float val) {
    // monotonic bit-pattern trick; addr initialized to -inf
    if (val >= 0.0f) atomicMax((int*)addr, __float_as_int(val));
    else             atomicMin((unsigned int*)addr, __float_as_uint(val));
}

// =====================================================================
// WMMA GEMM:  out[M,Nc] = bf16(A[M,K]) @ W + bias
//   A : bf16 row-major [M,K]
//   Wpk : bf16 pre-packed in WMMA B-fragment order:
//         element (tn, kbi, lane, j)  at  ((tn*KB + kbi)*32 + lane)*16 + j
//         holds W[kbi*32 + 16*(j>>3) + 8*(lane>>4) + (j&7) ,  tn*16 + (lane&15)]
//   Each wave computes a 64x16 strip (4 M-tiles), so the B fragment is
//   loaded once per K-step and fed to 4 back-to-back WMMAs.
// =====================================================================
#define GEMM_WAVES 4
#define MT 4
template<int K, bool OUT_BF16>
__global__ void gemm_bf16_wmma(const unsigned short* __restrict__ A,
                               const unsigned short* __restrict__ Wpk,
                               const float* __restrict__ bias,
                               void* __restrict__ outp,
                               int M, int Ncols)
{
    constexpr int KB = K / 32;
    const int lane = threadIdx.x & 31;
    const int wave = threadIdx.x >> 5;
    const int tilesN = Ncols >> 4;
    const int totalTiles = (M >> 6) * tilesN;      // 64-row strips
    const int tile = blockIdx.x * GEMM_WAVES + wave;
    if (tile >= totalTiles) return;                // uniform per wave -> EXEC all-ones
    const int tm = tile / tilesN;
    const int tn = tile - tm * tilesN;
    const int half = lane >> 4;
    const int l16  = lane & 15;

    const unsigned short* Ar[MT];
#pragma unroll
    for (int mt = 0; mt < MT; ++mt)
        Ar[mt] = A + (size_t)(tm * 64 + mt * 16 + l16) * K + 8 * half;

    const unsigned short* Bp = Wpk + (size_t)tn * KB * 512 + lane * 16;

    v8f acc[MT];
#pragma unroll
    for (int mt = 0; mt < MT; ++mt) acc[mt] = (v8f){};

#pragma unroll 2
    for (int kbi = 0; kbi < KB; ++kbi) {
        if (kbi + 1 < KB)
            __builtin_prefetch(Bp + (size_t)(kbi + 1) * 512, 0, 1);  // global_prefetch
        union { v16bf v; uint4 q[2]; } bu;
        const uint4* bp = (const uint4*)(Bp + (size_t)kbi * 512);
        bu.q[0] = bp[0];
        bu.q[1] = bp[1];
#pragma unroll
        for (int mt = 0; mt < MT; ++mt) {
            union { v16bf v; uint4 q[2]; } au;
            au.q[0] = *(const uint4*)(Ar[mt] + kbi * 32);
            au.q[1] = *(const uint4*)(Ar[mt] + kbi * 32 + 16);
            acc[mt] = __builtin_amdgcn_wmma_f32_16x16x32_bf16(false, au.v, false, bu.v,
                                                              (short)0, acc[mt], false, false);
        }
    }
    // C/D layout: element r -> M = r + 8*half, N = lane%16
#pragma unroll
    for (int mt = 0; mt < MT; ++mt) {
        const int ncol = tn * 16 + l16;
        const float b = bias[ncol];
#pragma unroll
        for (int r = 0; r < 8; ++r) {
            const int mrow = tm * 64 + mt * 16 + 8 * half + r;
            const float val = acc[mt][r] + b;
            if (OUT_BF16) ((unsigned short*)outp)[(size_t)mrow * Ncols + ncol] = f2bf(val);
            else          ((float*)outp)[(size_t)mrow * Ncols + ncol] = val;
        }
    }
}

// ---------------- weight packing into B-fragment order ----------------
// idx decode: j = idx&15, lane = (idx>>4)&31, rest = idx>>9 = tn*KB + kbi
__device__ __forceinline__ void frag_decode(int idx, int KB, int Ncols,
                                            int& srcIdx)
{
    const int j    = idx & 15;
    const int lane = (idx >> 4) & 31;
    const int rest = idx >> 9;
    const int kbi  = rest % KB;
    const int tn   = rest / KB;
    const int col  = tn * 16 + (lane & 15);
    const int kk   = kbi * 32 + ((j >> 3) << 4) + ((lane >> 4) << 3) + (j & 7);
    srcIdx = kk * Ncols + col;
}

__global__ void pack_frag_f32(const float* __restrict__ W, unsigned short* __restrict__ Wpk,
                              int K, int Ncols)
{
    int idx = blockIdx.x * blockDim.x + threadIdx.x;
    if (idx >= K * Ncols) return;
    int s; frag_decode(idx, K / 32, Ncols, s);
    Wpk[idx] = f2bf(W[s]);
}

__global__ void pack_frag_bf16(const unsigned short* __restrict__ W, unsigned short* __restrict__ Wpk,
                               int K, int Ncols)
{
    int idx = blockIdx.x * blockDim.x + threadIdx.x;
    if (idx >= K * Ncols) return;
    int s; frag_decode(idx, K / 32, Ncols, s);
    Wpk[idx] = W[s];
}

// ---------------- small prep kernels ----------------
__global__ void pack_wcat(const float* __restrict__ Wq, const float* __restrict__ Wk,
                          const float* __restrict__ Wv, const float* __restrict__ Wskip,
                          const float* __restrict__ bq, const float* __restrict__ bk,
                          const float* __restrict__ bv, const float* __restrict__ bskip,
                          unsigned short* __restrict__ Wcat, float* __restrict__ bcat)
{
    int idx = blockIdx.x * blockDim.x + threadIdx.x;
    if (idx < DD * CAT) {
        int row = idx / CAT, col = idx - row * CAT;
        float v;
        if      (col < 256) v = Wq[row * 256 + col];
        else if (col < 512) v = Wk[row * 256 + (col - 256)];
        else if (col < 768) v = Wv[row * 256 + (col - 512)];
        else                v = Wskip[row * 128 + (col - 768)];
        Wcat[idx] = f2bf(v);
    }
    if (idx < CAT) {
        float b;
        if      (idx < 256) b = bq[idx];
        else if (idx < 512) b = bk[idx - 256];
        else if (idx < 768) b = bv[idx - 512];
        else                b = bskip[idx - 768];
        bcat[idx] = b;
    }
}

__global__ void f32_to_bf16(const float* __restrict__ src, unsigned short* __restrict__ dst, int n)
{
    int i = blockIdx.x * blockDim.x + threadIdx.x;
    if (i < n) dst[i] = f2bf(src[i]);
}

__global__ void fill_neginf(float* __restrict__ p, int n)
{
    int i = blockIdx.x * blockDim.x + threadIdx.x;
    if (i < n) p[i] = __int_as_float(0xFF800000);
}

// hetro_proj[t, o] = sum_d hetro_emb[t,d] * We[d,o]    (10x256, exact fp32)
__global__ void hetro_proj_kernel(const float* __restrict__ emb, const float* __restrict__ We,
                                  float* __restrict__ hp)
{
    int idx = blockIdx.x * blockDim.x + threadIdx.x;
    if (idx >= NTYPES * HC) return;
    int t = idx / HC, o = idx - t * HC;
    float s = 0.f;
#pragma unroll 8
    for (int d = 0; d < DD; ++d) s += emb[t * DD + d] * We[d * HC + o];
    hp[idx] = s;
}

// ---------------- edge pass 1: logits + segment max ----------------
__global__ void edge_logits(const int* __restrict__ eidx, const int* __restrict__ etype,
                            const float* __restrict__ qkvs, const float* __restrict__ hproj,
                            float* __restrict__ la, float* __restrict__ nmax)
{
    const int lane = threadIdx.x & 31;
    const int gw = blockIdx.x * (blockDim.x >> 5) + (threadIdx.x >> 5);
    if (gw >= EE * HH) return;
    const int e  = gw >> 1;
    const int hh = gw & 1;
    const int src = eidx[e];
    const int dst = eidx[EE + e];
    const int t   = etype[e];
    const float* qp = qkvs + (size_t)dst * CAT + hh * CC;            // q
    const float* kp = qkvs + (size_t)src * CAT + HC + hh * CC;       // k
    const float* ep = hproj + t * HC + hh * CC;
    float s = 0.f;
#pragma unroll
    for (int c = lane; c < CC; c += 32) s += qp[c] * (kp[c] + ep[c]);
#pragma unroll
    for (int off = 16; off > 0; off >>= 1) s += __shfl_xor(s, off, 32);
    if (lane == 0) {
        s *= 0.08838834764831845f;   // 1/sqrt(128)
        la[e * HH + hh] = s;
        atomicMaxFloat(&nmax[dst * HH + hh], s);
    }
}

// ---------------- edge pass 2: exp + segment sum ----------------
__global__ void edge_expsum(const int* __restrict__ eidx, float* __restrict__ la,
                            const float* __restrict__ nmax, float* __restrict__ nsum)
{
    int idx = blockIdx.x * blockDim.x + threadIdx.x;
    if (idx >= EE * HH) return;
    const int e  = idx >> 1;
    const int hh = idx & 1;
    const int dst = eidx[EE + e];
    float a = __expf(la[idx] - nmax[dst * HH + hh]);
    la[idx] = a;
    atomicAdd(&nsum[dst * HH + hh], a);
}

// ---------------- edge pass 3: weighted scatter-add of messages ----------------
__global__ void edge_scatter(const int* __restrict__ eidx, const int* __restrict__ etype,
                             const float* __restrict__ qkvs, const float* __restrict__ hproj,
                             const float* __restrict__ la, const float* __restrict__ nsum,
                             float* __restrict__ agg)
{
    int idx = blockIdx.x * blockDim.x + threadIdx.x;     // e*64 + hh*32 + cg
    if (idx >= EE * HH * 32) return;
    const int cg = idx & 31;
    const int hh = (idx >> 5) & 1;
    const int e  = idx >> 6;
    const int src = eidx[e];
    const int dst = eidx[EE + e];
    const int t   = etype[e];
    const float alpha = la[e * HH + hh] / (nsum[dst * HH + hh] + 1e-16f);
    const int c0 = cg * 4;
    const float* vp = qkvs + (size_t)src * CAT + 2 * HC + hh * CC + c0;  // v
    const float* ep = hproj + t * HC + hh * CC + c0;
    float* ag = agg + ((size_t)dst * HH + hh) * CC + c0;
#pragma unroll
    for (int j = 0; j < 4; ++j) atomicAdd(&ag[j], (vp[j] + ep[j]) * alpha);
}

// ---------------- head-mean + skip + residual + LayerNorm (-> bf16) ----------------
__global__ void node_post(const float* __restrict__ agg, const float* __restrict__ qkvs,
                          const float* __restrict__ pre, const float* __restrict__ gamma,
                          const float* __restrict__ beta, unsigned short* __restrict__ hlnbf)
{
    const int n = blockIdx.x;
    const int c = threadIdx.x;
    float outv = 0.5f * (agg[((size_t)n * 2 + 0) * CC + c] + agg[((size_t)n * 2 + 1) * CC + c])
               + qkvs[(size_t)n * CAT + 768 + c]
               + pre[(size_t)n * DD + c];
    __shared__ float buf[DD];
    buf[c] = outv; __syncthreads();
    for (int s = 64; s > 0; s >>= 1) { if (c < s) buf[c] += buf[c + s]; __syncthreads(); }
    const float mu = buf[0] * (1.0f / DD);
    __syncthreads();
    const float dv = outv - mu;
    buf[c] = dv * dv; __syncthreads();
    for (int s = 64; s > 0; s >>= 1) { if (c < s) buf[c] += buf[c + s]; __syncthreads(); }
    const float var = buf[0] * (1.0f / DD);
    const float y = dv * rsqrtf(var + 1e-6f) * gamma[c] + beta[c];
    hlnbf[(size_t)n * DD + c] = f2bf(y);
}

// ---------------- final gather ----------------
__global__ void gather_out(const float* __restrict__ h, const int* __restrict__ mask,
                           float* __restrict__ out)
{
    int idx = blockIdx.x * blockDim.x + threadIdx.x;
    if (idx >= BB * DD) return;
    int b = idx / DD, c = idx - b * DD;
    out[idx] = h[(size_t)mask[b] * DD + c];
}

// ---------------- host launcher ----------------
static inline int cdiv(long long a, long long b) { return (int)((a + b - 1) / b); }

extern "C" void kernel_launch(void* const* d_in, const int* in_sizes, int n_in,
                              void* d_out, int out_size, void* d_ws, size_t ws_size,
                              hipStream_t stream)
{
    const float* x        = (const float*)d_in[0];
    const float* hemb     = (const float*)d_in[1];
    const float* Wq       = (const float*)d_in[2];
    const float* bq       = (const float*)d_in[3];
    const float* Wk       = (const float*)d_in[4];
    const float* bk       = (const float*)d_in[5];
    const float* Wv       = (const float*)d_in[6];
    const float* bv       = (const float*)d_in[7];
    const float* We       = (const float*)d_in[8];
    const float* Wskip    = (const float*)d_in[9];
    const float* bskip    = (const float*)d_in[10];
    const float* ln_gamma = (const float*)d_in[11];
    const float* ln_beta  = (const float*)d_in[12];
    const float* ffn_w1   = (const float*)d_in[13];
    const float* ffn_b1   = (const float*)d_in[14];
    const float* ffn_w2   = (const float*)d_in[15];
    const float* ffn_b2   = (const float*)d_in[16];
    const int*   eidx     = (const int*)d_in[17];
    const int*   etype    = (const int*)d_in[18];
    const int*   mask     = (const int*)d_in[19];
    float* out = (float*)d_out;

    // workspace carve-up (256B aligned)
    char* ws = (char*)d_ws;
    auto carve = [&](size_t bytes) { char* p = ws; ws += (bytes + 255) & ~(size_t)255; return p; };
    unsigned short* hbf     = (unsigned short*)carve((size_t)NN * DD * 2);
    unsigned short* Wcat    = (unsigned short*)carve((size_t)DD * CAT * 2);   // row-major temp
    float*          bcat    = (float*)carve((size_t)CAT * 4);
    unsigned short* WcatPk  = (unsigned short*)carve((size_t)DD * CAT * 2);   // fragment order
    unsigned short* Wff1Pk  = (unsigned short*)carve((size_t)LL * DD * II * 2);
    unsigned short* Wff2Pk  = (unsigned short*)carve((size_t)LL * II * DD * 2);
    float*          hproj   = (float*)carve((size_t)NTYPES * HC * 4);
    float*          qkvs    = (float*)carve((size_t)NN * CAT * 4);
    float*          la      = (float*)carve((size_t)EE * HH * 4);
    float*          nmax    = (float*)carve((size_t)NN * HH * 4);
    float*          nsum    = (float*)carve((size_t)NN * HH * 4);
    float*          agg     = (float*)carve((size_t)NN * HH * CC * 4);
    unsigned short* hlnbf   = (unsigned short*)carve((size_t)NN * DD * 2);
    unsigned short* hmidbf  = (unsigned short*)carve((size_t)NN * II * 2);
    float*          hcur    = (float*)carve((size_t)NN * DD * 4);

    // one-time prep: concat -> bf16, then repack into WMMA B-fragment order
    pack_wcat<<<cdiv(DD * CAT, 256), 256, 0, stream>>>(Wq, Wk, Wv, Wskip, bq, bk, bv, bskip, Wcat, bcat);
    pack_frag_bf16<<<cdiv(DD * CAT, 256), 256, 0, stream>>>(Wcat, WcatPk, DD, CAT);
    for (int l = 0; l < LL; ++l) {
        pack_frag_f32<<<cdiv(DD * II, 256), 256, 0, stream>>>(ffn_w1 + (size_t)l * DD * II,
                                                              Wff1Pk + (size_t)l * DD * II, DD, II);
        pack_frag_f32<<<cdiv(II * DD, 256), 256, 0, stream>>>(ffn_w2 + (size_t)l * II * DD,
                                                              Wff2Pk + (size_t)l * II * DD, II, DD);
    }
    hetro_proj_kernel<<<cdiv(NTYPES * HC, 256), 256, 0, stream>>>(hemb, We, hproj);

    for (int l = 0; l < LL; ++l) {
        const float* hsrc = (l == 0) ? x : hcur;

        // h -> bf16
        f32_to_bf16<<<cdiv((long long)NN * DD, 256), 256, 0, stream>>>(hsrc, hbf, NN * DD);

        // fused q|k|v|skip GEMM: [N,128] @ [128,896]
        {
            int tiles = (NN / 64) * (CAT / 16);
            gemm_bf16_wmma<DD, false><<<cdiv(tiles, GEMM_WAVES), GEMM_WAVES * 32, 0, stream>>>(
                hbf, WcatPk, bcat, qkvs, NN, CAT);
        }

        // init segment buffers
        hipMemsetAsync(nsum, 0, (size_t)NN * HH * 4, stream);
        hipMemsetAsync(agg,  0, (size_t)NN * HH * CC * 4, stream);
        fill_neginf<<<cdiv(NN * HH, 256), 256, 0, stream>>>(nmax, NN * HH);

        // edge passes
        edge_logits<<<cdiv((long long)EE * HH, 8), 256, 0, stream>>>(eidx, etype, qkvs, hproj, la, nmax);
        edge_expsum<<<cdiv((long long)EE * HH, 256), 256, 0, stream>>>(eidx, la, nmax, nsum);
        edge_scatter<<<cdiv((long long)EE * HH * 32, 256), 256, 0, stream>>>(eidx, etype, qkvs, hproj, la, nsum, agg);

        // head mean + skip + residual + LN -> bf16
        node_post<<<NN, DD, 0, stream>>>(agg, qkvs, hsrc,
                                         ln_gamma + (size_t)l * DD, ln_beta + (size_t)l * DD, hlnbf);

        // FFN: [N,128]@[128,512] -> bf16, then [N,512]@[512,128] -> f32 hcur
        {
            int tiles1 = (NN / 64) * (II / 16);
            gemm_bf16_wmma<DD, true><<<cdiv(tiles1, GEMM_WAVES), GEMM_WAVES * 32, 0, stream>>>(
                hlnbf, Wff1Pk + (size_t)l * DD * II, ffn_b1 + (size_t)l * II, hmidbf, NN, II);
            int tiles2 = (NN / 64) * (DD / 16);
            gemm_bf16_wmma<II, false><<<cdiv(tiles2, GEMM_WAVES), GEMM_WAVES * 32, 0, stream>>>(
                hmidbf, Wff2Pk + (size_t)l * II * DD, ffn_b2 + (size_t)l * DD, hcur, NN, DD);
        }
    }

    gather_out<<<cdiv(BB * DD, 256), 256, 0, stream>>>(hcur, mask, out);
}
